// MLAttention_51548197486850
// MI455X (gfx1250) — compile-verified, matrix operand order checked by reading, not run on hardware
//
#include <hip/hip_runtime.h>
#include <hip/hip_bf16.h>
#include <stdint.h>

// ---- problem constants (from reference) ----
#define NHEAD 16
#define HD    128
#define DRN   32
#define DNN   96
#define HDIM  2048
#define BBATCH 2
#define LSEQ  2048
#define MROWS (BBATCH*LSEQ)   // 4096

typedef __attribute__((ext_vector_type(16))) __bf16       v16bf;
typedef __attribute__((ext_vector_type(8)))  float        v8f;
typedef __attribute__((ext_vector_type(8)))  unsigned int v8u;
typedef __attribute__((ext_vector_type(4)))  unsigned int v4u;
typedef __attribute__((ext_vector_type(8)))  int          v8i;
typedef __attribute__((ext_vector_type(4)))  int          v4i;

// fp32 -> bf16 round-to-nearest-even
__device__ __forceinline__ unsigned short f2bf(float f) {
  unsigned int u = __builtin_bit_cast(unsigned int, f);
  u += 0x7FFFu + ((u >> 16) & 1u);
  return (unsigned short)(u >> 16);
}
__device__ __forceinline__ float bf2f(unsigned short u) {
  unsigned int x = ((unsigned int)u) << 16;
  return __builtin_bit_cast(float, x);
}
// build 16-half fragment from two 16B chunks (K=8g..8g+7 and K=16+8g..16+8g+7)
__device__ __forceinline__ v16bf frag(v4u lo, v4u hi) {
  v8u u = __builtin_shufflevector(lo, hi, 0, 1, 2, 3, 4, 5, 6, 7);
  return __builtin_bit_cast(v16bf, u);
}

// ------------------------------------------------------------------
// TDM: DMA a 64-row x 32-element (bf16) 2D tile global->LDS.
// D# per CDNA5 ISA ch.8: group0 {count=1, lds_addr, global_addr, type=2},
// group1 {data_size=2B, tensor/tile dims, dim0 stride}, groups 2/3 zero.
// All fields are workgroup-uniform -> SGPR descriptor groups.
// (6-arg builtin form: g0, g1, g2, g3, g4, cpol)
// ------------------------------------------------------------------
__device__ __forceinline__ void tdm_load_tile64x32(unsigned int lds_addr,
                                                   const unsigned short* gp,
                                                   int stride_el) {
  unsigned long long ga = (unsigned long long)gp;
  v4u g0;
  g0[0] = 1u;                                             // count=1
  g0[1] = lds_addr;                                       // lds byte addr
  g0[2] = (unsigned int)(ga & 0xFFFFFFFFull);             // global_addr[31:0]
  g0[3] = (unsigned int)((ga >> 32) & 0x01FFFFFFull)      // global_addr[56:32]
          | (2u << 30);                                   // type=2 ("image")
  v8i g1;
  g1[0] = (int)(1u << 16);          // wg_mask=0, data_size=1 (2 bytes)
  g1[1] = (int)(32u << 16);         // tensor_dim0[15:0]=32 (bits 63:48)
  g1[2] = (int)(64u << 16);         // tensor_dim0[31:16]=0 | tensor_dim1[15:0]=64
  g1[3] = (int)(32u << 16);         // tensor_dim1[31:16]=0 | tile_dim0=32
  g1[4] = (int)64;                  // tile_dim1=64 | tile_dim2=0
  g1[5] = (int)stride_el;           // tensor_dim0_stride[31:0]
  g1[6] = 0;                        // dim0_stride[47:32]=0 | dim1_stride[15:0]=0
  g1[7] = 0;                        // dim1_stride[47:16]=0
  v4i g2 = {0, 0, 0, 0};
  v4i g3 = {0, 0, 0, 0};
  v8i g4 = {0, 0, 0, 0, 0, 0, 0, 0};
  __builtin_amdgcn_tensor_load_to_lds(g0, g1, g2, g3, g4, 0);
}

// ------------------------------------------------------------------
// fp32 -> bf16 elementwise conversion
// ------------------------------------------------------------------
__global__ __launch_bounds__(256)
void cvt_bf16(const float* __restrict__ in, unsigned short* __restrict__ out, int n) {
  int i = blockIdx.x * 256 + threadIdx.x;
  if (i < n) out[i] = f2bf(in[i]);
}

// fp32 [K][N] -> bf16 transposed [N][K]
__global__ __launch_bounds__(256)
void cvt_bf16_T(const float* __restrict__ in, unsigned short* __restrict__ out,
                int K, int N) {
  int i = blockIdx.x * 256 + threadIdx.x;
  if (i >= K * N) return;
  int k = i / N, n = i - k * N;
  out[(size_t)n * K + k] = f2bf(in[i]);
}

// ------------------------------------------------------------------
// C[M,N] = A[M,K] @ B[K,N] with B given transposed (BT = [N][K], K-contig).
// A,BT bf16; C fp32 or bf16. Block 128 threads = 4 waves; tile 64x64.
// Tiles staged by TDM (tensor_load_to_lds), double-buffered LDS,
// fragments loaded as paired ds_load_b128.
// ------------------------------------------------------------------
template <bool OUT_F32>
__global__ __launch_bounds__(128)
void gemm_bf16(const unsigned short* __restrict__ A,
               const unsigned short* __restrict__ BT,
               void* __restrict__ Cv, int M, int N, int K) {
  __shared__ __align__(16) unsigned short As[2][64*32];   // [buf][row][k]
  __shared__ __align__(16) unsigned short Bs[2][64*32];   // [buf][n][k]
  (void)M;
  const int t    = threadIdx.x;
  const int w    = t >> 5;
  const int lane = t & 31;
  const int g    = lane >> 4;
  const int ln   = lane & 15;
  const int tM   = blockIdx.y * 64;
  const int tN   = blockIdx.x * 64;
  const bool issuer = (t < 32);   // wave 0 drives the TDM

  const unsigned short* Abase = &A[(size_t)tM * K];
  const unsigned short* Bbase = &BT[(size_t)tN * K];

  if (issuer) {
    tdm_load_tile64x32((unsigned int)(size_t)&As[0][0], Abase, K);
    tdm_load_tile64x32((unsigned int)(size_t)&Bs[0][0], Bbase, K);
    __builtin_amdgcn_s_wait_tensorcnt(0);
  }
  __syncthreads();

  v8f acc[4] = {};
  int buf = 0;
  for (int kb = 0; kb < K; kb += 32) {
    const bool has_next = (kb + 32) < K;
    if (has_next && issuer) {
      tdm_load_tile64x32((unsigned int)(size_t)&As[buf^1][0], Abase + kb + 32, K);
      tdm_load_tile64x32((unsigned int)(size_t)&Bs[buf^1][0], Bbase + kb + 32, K);
    }

    // A fragment for this wave's 16 rows: two ds_load_b128
    const unsigned short* arow = &As[buf][(w*16 + ln)*32 + 8*g];
    v16bf af = frag(((const v4u*)arow)[0], ((const v4u*)arow)[2]);
    #pragma unroll
    for (int nt = 0; nt < 4; ++nt) {
      const unsigned short* brow = &Bs[buf][(nt*16 + ln)*32 + 8*g];
      v16bf bfr = frag(((const v4u*)brow)[0], ((const v4u*)brow)[2]);
      acc[nt] = __builtin_amdgcn_wmma_f32_16x16x32_bf16(
          false, af, false, bfr, (short)0, acc[nt], false, false);
    }

    if (has_next && issuer) __builtin_amdgcn_s_wait_tensorcnt(0);
    __syncthreads();
    buf ^= 1;
  }

  #pragma unroll
  for (int nt = 0; nt < 4; ++nt)
    #pragma unroll
    for (int r = 0; r < 8; ++r) {
      int m = r + 8*g;
      size_t idx = (size_t)(tM + w*16 + m)*N + tN + nt*16 + ln;
      if (OUT_F32) ((float*)Cv)[idx] = acc[nt][r];
      else         ((unsigned short*)Cv)[idx] = f2bf(acc[nt][r]);
    }
}

// ------------------------------------------------------------------
// RoPE + head-major packing (bf16 in, bf16 out).
//   qf[(b,h,l,d)], kf[(b,h,l,d)] = concat(nope, rope(rope_part))
//   vT[(b,h,d,l)] = v transposed per head
// ------------------------------------------------------------------
__global__ __launch_bounds__(256)
void rope_pack(const unsigned short* __restrict__ q,
               const unsigned short* __restrict__ knope,
               const unsigned short* __restrict__ krope,
               const unsigned short* __restrict__ v,
               const float* __restrict__ cost, const float* __restrict__ sint,
               unsigned short* __restrict__ qf, unsigned short* __restrict__ kf,
               unsigned short* __restrict__ vT) {
  int gid = blockIdx.x * 256 + threadIdx.x;      // 0 .. 2^23-1
  int d = gid & 127;
  int h = (gid >> 7) & 15;
  int l = (gid >> 11) & 2047;
  int b = gid >> 22;
  int row = b*LSEQ + l;

  float qv, kv;
  if (d < DNN) {
    qv = bf2f(q[(size_t)row*HDIM + h*HD + d]);
    kv = bf2f(knope[(size_t)row*(NHEAD*DNN) + h*DNN + d]);
  } else {
    int dr = d - DNN;
    float c = cost[l*DRN + dr], s = sint[l*DRN + dr];
    const unsigned short* qr = &q[(size_t)row*HDIM + h*HD + DNN];
    const unsigned short* kr = &krope[(size_t)row*(NHEAD*DRN) + h*DRN];
    float rotq = (dr < 16) ? -bf2f(qr[dr + 16]) : bf2f(qr[dr - 16]);
    float rotk = (dr < 16) ? -bf2f(kr[dr + 16]) : bf2f(kr[dr - 16]);
    qv = bf2f(qr[dr])*c + rotq*s;
    kv = bf2f(kr[dr])*c + rotk*s;
  }
  float vv = bf2f(v[(size_t)row*HDIM + h*HD + d]);

  int bh = b*NHEAD + h;
  qf[((size_t)bh*LSEQ + l)*HD + d] = f2bf(qv);
  kf[((size_t)bh*LSEQ + l)*HD + d] = f2bf(kv);
  vT[((size_t)bh*HD + d)*LSEQ + l] = f2bf(vv);
}

// ------------------------------------------------------------------
// Causal flash attention, bf16 WMMA. Block 128 = 4 waves; each wave
// owns a 16-query tile; 32-key blocks with online softmax.
// out: bf16, (B, L, NH*HD) layout, feeds the Wo GEMM directly.
// ------------------------------------------------------------------
__global__ __launch_bounds__(128)
void flash_attn(const unsigned short* __restrict__ qf,
                const unsigned short* __restrict__ kf,
                const unsigned short* __restrict__ vT,
                unsigned short* __restrict__ out) {
  __shared__ __align__(16) unsigned short Pl[4*16*32];  // per-wave P tile
  const int w    = threadIdx.x >> 5;
  const int lane = threadIdx.x & 31;
  const int g    = lane >> 4;
  const int ln   = lane & 15;
  const int bh   = blockIdx.y;             // b*16 + h
  const int b    = bh >> 4, h = bh & 15;
  const int qbase = blockIdx.x*64 + w*16;
  const float scale = 0.08838834764831845f; // 1/sqrt(128)

  // Q A-fragments (4 K-chunks of 32 over d=128): paired b128 loads
  v16bf qa[4];
  {
    const v4u* qrow = (const v4u*)&qf[((size_t)bh*LSEQ + qbase + ln)*HD];
    #pragma unroll
    for (int c = 0; c < 4; ++c)
      qa[c] = frag(qrow[4*c + g], qrow[4*c + g + 2]);
  }

  v8f o[8] = {};
  float rmax[8], rsum[8];
  #pragma unroll
  for (int r = 0; r < 8; ++r) { rmax[r] = -3.0e38f; rsum[r] = 0.0f; }

  const int nkb = (qbase + 15)/32 + 1;
  for (int kb = 0; kb < nkb; ++kb) {
    const int kbase = kb*32;
    const bool need_mask = (kbase + 31) > qbase;

    // S = Q @ K^T for 32 keys (two 16-key subtiles)
    v8f s0 = {}, s1 = {};
    {
      const v4u* kr0 = (const v4u*)&kf[((size_t)bh*LSEQ + kbase + ln)*HD];
      const v4u* kr1 = (const v4u*)&kf[((size_t)bh*LSEQ + kbase + 16 + ln)*HD];
      #pragma unroll
      for (int c = 0; c < 4; ++c) {
        v16bf b0 = frag(kr0[4*c + g], kr0[4*c + g + 2]);
        v16bf b1 = frag(kr1[4*c + g], kr1[4*c + g + 2]);
        s0 = __builtin_amdgcn_wmma_f32_16x16x32_bf16(
            false, qa[c], false, b0, (short)0, s0, false, false);
        s1 = __builtin_amdgcn_wmma_f32_16x16x32_bf16(
            false, qa[c], false, b1, (short)0, s1, false, false);
      }
    }

    // scale + causal mask (C layout: row m = r + 8g, col n = ln)
    float t0[8], t1[8];
    #pragma unroll
    for (int r = 0; r < 8; ++r) {
      int m = r + 8*g;
      float a0 = s0[r]*scale, a1 = s1[r]*scale;
      if (need_mask) {
        if (kbase + ln      > qbase + m) a0 = -1.0e9f;
        if (kbase + 16 + ln > qbase + m) a1 = -1.0e9f;
      }
      t0[r] = a0; t1[r] = a1;
    }

    // row max over 32-key block (reduce over 16 lanes of this group)
    float bm[8];
    #pragma unroll
    for (int r = 0; r < 8; ++r) bm[r] = fmaxf(t0[r], t1[r]);
    #pragma unroll
    for (int off = 1; off < 16; off <<= 1)
      #pragma unroll
      for (int r = 0; r < 8; ++r)
        bm[r] = fmaxf(bm[r], __shfl_xor(bm[r], off, 32));

    float alpha[8];
    #pragma unroll
    for (int r = 0; r < 8; ++r) {
      float nm = fmaxf(rmax[r], bm[r]);
      alpha[r] = __expf(rmax[r] - nm);
      rmax[r]  = nm;
      t0[r] = __expf(t0[r] - nm);
      t1[r] = __expf(t1[r] - nm);
    }

    // row sum of P
    float ps[8];
    #pragma unroll
    for (int r = 0; r < 8; ++r) ps[r] = t0[r] + t1[r];
    #pragma unroll
    for (int off = 1; off < 16; off <<= 1)
      #pragma unroll
      for (int r = 0; r < 8; ++r)
        ps[r] += __shfl_xor(ps[r], off, 32);
    #pragma unroll
    for (int r = 0; r < 8; ++r) rsum[r] = rsum[r]*alpha[r] + ps[r];

    // rescale O
    #pragma unroll
    for (int nt = 0; nt < 8; ++nt)
      #pragma unroll
      for (int r = 0; r < 8; ++r)
        o[nt][r] *= alpha[r];

    // P (C layout) -> LDS row-major 16x32 bf16 -> reload as A fragment
    unsigned short* pw = &Pl[w*16*32];
    #pragma unroll
    for (int r = 0; r < 8; ++r) {
      int m = r + 8*g;
      pw[m*32 + ln]      = f2bf(t0[r]);
      pw[m*32 + 16 + ln] = f2bf(t1[r]);
    }
    asm volatile("s_wait_dscnt 0" ::: "memory");   // cross-lane LDS RAW
    const unsigned short* prd = pw + ln*32 + 8*g;
    v16bf pa = frag(((const v4u*)prd)[0], ((const v4u*)prd)[2]);

    // O += P @ V  (V^T: K-pairs contiguous -> paired b128 loads)
    #pragma unroll
    for (int nt = 0; nt < 8; ++nt) {
      const v4u* vcol =
          (const v4u*)&vT[((size_t)bh*HD + nt*16 + ln)*LSEQ + kbase];
      v16bf vb = frag(vcol[g], vcol[g + 2]);
      o[nt] = __builtin_amdgcn_wmma_f32_16x16x32_bf16(
          false, pa, false, vb, (short)0, o[nt], false, false);
    }
  }

  // normalize + store bf16 (B, L, NH*HD)
  #pragma unroll
  for (int nt = 0; nt < 8; ++nt)
    #pragma unroll
    for (int r = 0; r < 8; ++r) {
      int m = r + 8*g;
      out[((size_t)(b*LSEQ + qbase + m))*HDIM + h*HD + nt*16 + ln] =
          f2bf(o[nt][r] / rsum[r]);
    }
}

// ------------------------------------------------------------------
extern "C" void kernel_launch(void* const* d_in, const int* in_sizes, int n_in,
                              void* d_out, int out_size, void* d_ws, size_t ws_size,
                              hipStream_t stream) {
  (void)in_sizes; (void)n_in; (void)out_size; (void)ws_size;
  const float* x    = (const float*)d_in[0];
  const float* Wq   = (const float*)d_in[1];
  const float* Wkd  = (const float*)d_in[2];
  const float* Wknu = (const float*)d_in[3];
  const float* Wvu  = (const float*)d_in[4];
  const float* Wkr  = (const float*)d_in[5];
  const float* Wo   = (const float*)d_in[6];
  const float* cost = (const float*)d_in[7];
  const float* sint = (const float*)d_in[8];

  char* ws = (char*)d_ws;
  const size_t MB = 1ull << 20;
  unsigned short* xb    = (unsigned short*)(ws +   0*MB); // 16MB
  unsigned short* WqT   = (unsigned short*)(ws +  16*MB); //  8MB  [2048][2048]
  unsigned short* WkdT  = (unsigned short*)(ws +  24*MB); //  2MB  [512][2048]
  unsigned short* WknuT = (unsigned short*)(ws +  26*MB); //  2MB  [1536][512]
  unsigned short* WvuT  = (unsigned short*)(ws +  28*MB); //  2MB  [2048][512]
  unsigned short* WkrT  = (unsigned short*)(ws +  30*MB); //  2MB  [512][2048]
  unsigned short* WoT   = (unsigned short*)(ws +  32*MB); //  8MB  [2048][2048]
  unsigned short* q_bf    = (unsigned short*)(ws +  40*MB); // 16MB
  unsigned short* ckv_bf  = (unsigned short*)(ws +  56*MB); //  4MB
  unsigned short* knope_bf= (unsigned short*)(ws +  60*MB); // 12MB
  unsigned short* krope_bf= (unsigned short*)(ws +  72*MB); //  4MB
  unsigned short* v_bf    = (unsigned short*)(ws +  76*MB); // 16MB
  unsigned short* qfb     = (unsigned short*)(ws +  92*MB); // 16MB
  unsigned short* kfb     = (unsigned short*)(ws + 108*MB); // 16MB
  unsigned short* vTb     = (unsigned short*)(ws + 124*MB); // 16MB
  unsigned short* attn_bf = (unsigned short*)(ws + 140*MB); // 16MB

  // ---- one-time conversions: x plain, weights transposed (for TDM B-tiles) ----
  cvt_bf16<<<dim3((MROWS*HDIM)/256), dim3(256), 0, stream>>>(x, xb, MROWS*HDIM);
  auto cvtT = [&](const float* src, unsigned short* dst, int K, int N) {
    cvt_bf16_T<<<dim3((K*N + 255)/256), dim3(256), 0, stream>>>(src, dst, K, N);
  };
  cvtT(Wq,   WqT,   2048, 2048);
  cvtT(Wkd,  WkdT,  2048,  512);
  cvtT(Wknu, WknuT,  512, 1536);
  cvtT(Wvu,  WvuT,   512, 2048);
  cvtT(Wkr,  WkrT,  2048,  512);
  cvtT(Wo,   WoT,   2048, 2048);

  dim3 blk(128);
  // projections (bf16 in, bf16 out)
  gemm_bf16<false><<<dim3(2048/64, MROWS/64), blk, 0, stream>>>(xb,     WqT,   q_bf,     MROWS, 2048, 2048);
  gemm_bf16<false><<<dim3( 512/64, MROWS/64), blk, 0, stream>>>(xb,     WkdT,  ckv_bf,   MROWS,  512, 2048);
  gemm_bf16<false><<<dim3( 512/64, MROWS/64), blk, 0, stream>>>(xb,     WkrT,  krope_bf, MROWS,  512, 2048);
  gemm_bf16<false><<<dim3(1536/64, MROWS/64), blk, 0, stream>>>(ckv_bf, WknuT, knope_bf, MROWS, 1536,  512);
  gemm_bf16<false><<<dim3(2048/64, MROWS/64), blk, 0, stream>>>(ckv_bf, WvuT,  v_bf,     MROWS, 2048,  512);
  // rope + head-major packing
  rope_pack<<<dim3((MROWS*NHEAD*HD)/256), dim3(256), 0, stream>>>(
      q_bf, knope_bf, krope_bf, v_bf, cost, sint, qfb, kfb, vTb);
  // causal flash attention
  flash_attn<<<dim3(LSEQ/64, BBATCH*NHEAD), blk, 0, stream>>>(qfb, kfb, vTb, attn_bf);
  // output projection (fp32 out)
  gemm_bf16<true><<<dim3(2048/64, MROWS/64), blk, 0, stream>>>(attn_bf, WoT, d_out, MROWS, 2048, 2048);
}
